// TaskRetrival_12713103197274
// MI455X (gfx1250) — matrix operand order repaired and checked by reading, not compiled, not run on hardware
//
#include <hip/hip_runtime.h>

// ---------------------------------------------------------------------------
// TaskRetrieval: task_emb = mean(x,0); cossim(memory, task_emb); top-32 rows.
// memory: 1,000,000 x 256 f32.  Output: 32 x 256 f32 (selected rows, sorted
// by descending cosine similarity, ties -> lower index, like jax top_k).
// ---------------------------------------------------------------------------

typedef __attribute__((ext_vector_type(16))) _Float16 v16h;
typedef __attribute__((ext_vector_type(8)))  float    v8f;

#define EPSF      1e-8f
#define NEG_BIG   (-3.402823466e+38f)

#define ROWS          1000000
#define COLS          256
#define TILES_TOTAL   62500      // ROWS / 16 exactly
#define TILES_PER_BLK 128        // 2048 rows per block
#define NBLOCKS       489        // ceil(62500 / 128)
#define NCAND         (NBLOCKS * 32)

// ---------------------------------------------------------------------------
// Kernel 0: task embedding (column means of x) + its L2 norm.
// ---------------------------------------------------------------------------
__global__ __launch_bounds__(256) void temb_kernel(const float* __restrict__ x,
                                                   float* __restrict__ temb,
                                                   float* __restrict__ tnorm) {
    const int k = threadIdx.x;           // 0..255 = column
    float s = 0.0f;
    for (int i = 0; i < 128; ++i) s += x[i * COLS + k];
    s *= (1.0f / 128.0f);
    temb[k] = s;

    __shared__ float red[256];
    red[k] = s * s;
    __syncthreads();
    for (int st = 128; st > 0; st >>= 1) {
        if (k < st) red[k] += red[k + st];
        __syncthreads();
    }
    if (k == 0) tnorm[0] = sqrtf(red[0]);
}

// ---------------------------------------------------------------------------
// Kernel 1: fused GEMV (WMMA f16->f32) + row norms + per-block top-32.
// Each wave owns 16-row tiles. A-tile: 16x32 slice of memory rows in the
// CDNA5 f16 A layout (lane<16: K=0..7,16..23; lane>=16: K=8..15,24..31).
// B: task-embedding K-slice replicated across all 16 columns, so after 8
// accumulating WMMAs every lane's v8f C holds dot[baseRow + r + 8*h].
// ---------------------------------------------------------------------------
__global__ __launch_bounds__(256) void cossim_kernel(const float* __restrict__ mem,
                                                     const float* __restrict__ temb,
                                                     const float* __restrict__ tnormp,
                                                     float* __restrict__ candVal,
                                                     int*   __restrict__ candIdx) {
    __shared__ float vals[TILES_PER_BLK * 16];   // 2048 cosines
    __shared__ float redV[256];
    __shared__ int   redI[256];

    const int tid  = threadIdx.x;
    const int lane = tid & 31;
    const int wave = tid >> 5;          // 0..7
    const int h    = lane >> 4;         // lane-half
    const int mrow = lane & 15;         // row-in-tile

    for (int i = tid; i < TILES_PER_BLK * 16; i += 256) vals[i] = NEG_BIG;

    const float tnorm = fmaxf(tnormp[0], EPSF);

    // Preload B (replicated-column task embedding) as f16, once per wave.
    // Lane-half h holds K = 8*h + s (s<8), 16 + 8*h + (s-8) (s>=8) within each 32-wide K chunk.
    v16h bmat[8];
    #pragma unroll
    for (int kt = 0; kt < 8; ++kt) {
        const float* tp = temb + kt * 32 + h * 8;
        #pragma unroll
        for (int s = 0; s < 16; ++s) bmat[kt][s] = (_Float16)tp[(s < 8) ? s : s + 8];
    }

    const int blockTile0 = blockIdx.x * TILES_PER_BLK;
    const int firstTile  = blockTile0 + wave * 16;

    for (int it = 0; it < 16; ++it) {
        const int tile = firstTile + it;
        if (tile >= TILES_TOTAL) break;            // uniform within the wave
        const int    localTile = wave * 16 + it;
        const size_t baseRow   = (size_t)tile * 16;
        const float* rowp      = mem + (baseRow + (size_t)mrow) * COLS;

        v8f   c  = {};
        float sq = 0.0f;

        #pragma unroll
        for (int kt = 0; kt < 8; ++kt) {
            const float4* p = (const float4*)(rowp + kt * 32 + 8 * h);        // K = 8h..8h+7
            const float4* q = (const float4*)(rowp + kt * 32 + 16 + 8 * h);   // K = 16+8h..
            const float4 p0 = p[0], p1 = p[1];
            const float4 q0 = q[0], q1 = q[1];

            v16h a;
            a[0]  = (_Float16)p0.x; a[1]  = (_Float16)p0.y;
            a[2]  = (_Float16)p0.z; a[3]  = (_Float16)p0.w;
            a[4]  = (_Float16)p1.x; a[5]  = (_Float16)p1.y;
            a[6]  = (_Float16)p1.z; a[7]  = (_Float16)p1.w;
            a[8]  = (_Float16)q0.x; a[9]  = (_Float16)q0.y;
            a[10] = (_Float16)q0.z; a[11] = (_Float16)q0.w;
            a[12] = (_Float16)q1.x; a[13] = (_Float16)q1.y;
            a[14] = (_Float16)q1.z; a[15] = (_Float16)q1.w;

            sq = fmaf(p0.x, p0.x, sq); sq = fmaf(p0.y, p0.y, sq);
            sq = fmaf(p0.z, p0.z, sq); sq = fmaf(p0.w, p0.w, sq);
            sq = fmaf(p1.x, p1.x, sq); sq = fmaf(p1.y, p1.y, sq);
            sq = fmaf(p1.z, p1.z, sq); sq = fmaf(p1.w, p1.w, sq);
            sq = fmaf(q0.x, q0.x, sq); sq = fmaf(q0.y, q0.y, sq);
            sq = fmaf(q0.z, q0.z, sq); sq = fmaf(q0.w, q0.w, sq);
            sq = fmaf(q1.x, q1.x, sq); sq = fmaf(q1.y, q1.y, sq);
            sq = fmaf(q1.z, q1.z, sq); sq = fmaf(q1.w, q1.w, sq);

            c = __builtin_amdgcn_wmma_f32_16x16x32_f16(
                    false, a, false, bmat[kt], (short)0, c, false, false);
        }

        // Full row sq-norm: lanes l and l^16 each saw half the K range.
        const float sqFull = sq + __shfl_xor(sq, 16, 32);
        const float rn     = fmaxf(sqrtf(sqFull), EPSF);

        // Lane holds dot[baseRow + r + 8h] in c[r]; fetch matching row norm.
        #pragma unroll
        for (int r = 0; r < 8; ++r) {
            const float rn_m = __shfl(rn, r + 8 * h, 32);
            const float cos  = c[r] / (rn_m * tnorm);
            if (mrow == 0) vals[localTile * 16 + r + 8 * h] = cos;
        }
    }
    __syncthreads();

    // Block-local top-32 (32 x argmax over 2048, tie -> lowest index).
    const size_t blockRowBase = (size_t)blockTile0 * 16;
    for (int sel = 0; sel < 32; ++sel) {
        float best = NEG_BIG; int bi = 0x7FFFFFFF;
        for (int i = tid; i < TILES_PER_BLK * 16; i += 256) {
            const float v = vals[i];
            if (v > best) { best = v; bi = i; }   // ascending i keeps lowest index
        }
        redV[tid] = best; redI[tid] = bi;
        __syncthreads();
        for (int s = 128; s > 0; s >>= 1) {
            if (tid < s) {
                const float v2 = redV[tid + s]; const int i2 = redI[tid + s];
                if (v2 > redV[tid] || (v2 == redV[tid] && i2 < redI[tid])) {
                    redV[tid] = v2; redI[tid] = i2;
                }
            }
            __syncthreads();
        }
        if (tid == 0) {
            const int w = redI[0];
            candVal[blockIdx.x * 32 + sel] = redV[0];
            candIdx[blockIdx.x * 32 + sel] = (int)(blockRowBase + (size_t)w);
            vals[w] = NEG_BIG;
        }
        __syncthreads();
    }
}

// ---------------------------------------------------------------------------
// Kernel 2: merge per-block top-32 lists -> global ordered top-32 indices.
// Tie-break on the *global row index* to match jax top_k ordering.
// ---------------------------------------------------------------------------
__global__ __launch_bounds__(256) void topk_merge(float* __restrict__ candVal,
                                                  const int* __restrict__ candIdx,
                                                  int* __restrict__ selIdx,
                                                  int nC) {
    __shared__ float redV[256];
    __shared__ int   redI[256];   // global row index
    __shared__ int   redP[256];   // candidate position (for marking)
    const int tid = threadIdx.x;

    for (int sel = 0; sel < 32; ++sel) {
        float best = NEG_BIG; int bIdx = 0x7FFFFFFF; int bPos = -1;
        for (int p = tid; p < nC; p += 256) {
            const float v = candVal[p];
            const int   g = candIdx[p];
            if (v > best || (v == best && g < bIdx)) { best = v; bIdx = g; bPos = p; }
        }
        redV[tid] = best; redI[tid] = bIdx; redP[tid] = bPos;
        __syncthreads();
        for (int s = 128; s > 0; s >>= 1) {
            if (tid < s) {
                const float v2 = redV[tid + s];
                const int   g2 = redI[tid + s];
                if (v2 > redV[tid] || (v2 == redV[tid] && g2 < redI[tid])) {
                    redV[tid] = v2; redI[tid] = g2; redP[tid] = redP[tid + s];
                }
            }
            __syncthreads();
        }
        if (tid == 0) {
            selIdx[sel] = redI[0];
            candVal[redP[0]] = NEG_BIG;
        }
        __syncthreads();
    }
}

// ---------------------------------------------------------------------------
// Kernel 3: gather the 32 selected rows (32 blocks x 256 threads).
// ---------------------------------------------------------------------------
__global__ __launch_bounds__(256) void gather_kernel(const float* __restrict__ mem,
                                                     const int* __restrict__ selIdx,
                                                     float* __restrict__ out) {
    const int j = blockIdx.x;     // output row 0..31
    const int t = threadIdx.x;    // column 0..255
    const size_t src = (size_t)selIdx[j] * COLS + (size_t)t;
    out[j * COLS + t] = mem[src];
}

// ---------------------------------------------------------------------------
extern "C" void kernel_launch(void* const* d_in, const int* in_sizes, int n_in,
                              void* d_out, int out_size, void* d_ws, size_t ws_size,
                              hipStream_t stream) {
    const float* x   = (const float*)d_in[0];   // 128 x 256 f32
    const float* mem = (const float*)d_in[1];   // 1e6 x 256 f32
    // d_in[2] = top_k (compile-time 32 here)
    float* out = (float*)d_out;                 // 32 x 256 f32

    char* ws = (char*)d_ws;
    float* temb    = (float*)(ws + 0);                        // 256 f32
    float* tnorm   = (float*)(ws + 1024);                     // 1 f32
    float* candVal = (float*)(ws + 2048);                     // NCAND f32
    int*   candIdx = (int*)  (ws + 2048 + NCAND * 4);         // NCAND i32
    int*   selIdx  = (int*)  (ws + 2048 + NCAND * 8);         // 32 i32

    temb_kernel  <<<1,       256, 0, stream>>>(x, temb, tnorm);
    cossim_kernel<<<NBLOCKS, 256, 0, stream>>>(mem, temb, tnorm, candVal, candIdx);
    topk_merge   <<<1,       256, 0, stream>>>(candVal, candIdx, selIdx, NCAND);
    gather_kernel<<<32,      256, 0, stream>>>(mem, selIdx, out);
}
